// Block_67748814127731
// MI455X (gfx1250) — matrix operand order repaired and checked by reading, not compiled, not run on hardware
//
#include <hip/hip_runtime.h>
#include <hip/hip_bf16.h>
#include <math.h>
#include <stdint.h>

// ---------------------------------------------------------------------------
// Types for CDNA5 WMMA
// ---------------------------------------------------------------------------
typedef __bf16 bf16x16 __attribute__((ext_vector_type(16)));
typedef float  f32x8   __attribute__((ext_vector_type(8)));

#define Bv   2
#define Tv   2048
#define Cv   1024
#define Hv   16
#define HSv  64
#define FFv  4096
#define M_ROWS (Bv * Tv)          // 4096
#define LDQKV  (3 * Cv)           // 3072

// GEMM output modes (compile-time -> no per-element branches)
#define MODE_BF16       0
#define MODE_F32_RESID  1
#define MODE_BF16_RELU  2

// ---------------------------------------------------------------------------
// CDNA5 async global->LDS copy (ASYNCcnt-tracked DMA; ISA 10.x / 15.18.3)
// 16 bytes per lane, LDS dest address per lane in a VGPR, no VGPR round-trip.
// ---------------------------------------------------------------------------
__device__ __forceinline__ void async_ld_b128(uint32_t lds_off,
                                              const __bf16* g) {
  asm volatile("global_load_async_to_lds_b128 %0, %1, off"
               :: "v"(lds_off), "v"(g) : "memory");
}
__device__ __forceinline__ void wait_async0() {
  asm volatile("s_wait_asynccnt 0x0" ::: "memory");
}

// ---------------------------------------------------------------------------
// Fragment loaders (wave32, 16x16x32 bf16 WMMA; this file's contiguous k order, used for A and B alike)
// A (16x32, 16-bit): lanes 0-15 row M=lane, K={0..15}; lanes 16-31 same
// rows, K={16..31}.  Two contiguous 16B loads per lane.
// ---------------------------------------------------------------------------
__device__ __forceinline__ bf16x16 load_a_frag(const __bf16* __restrict__ A,
                                               int lda, int lane) {
  const int r  = lane & 15;
  const int kg = lane >> 4;           // 0 or 1
  const __bf16* p = A + (size_t)r * lda + kg * 16;
  bf16x16 a;
#pragma unroll
  for (int e = 0; e < 8; ++e) a[e] = p[e];
#pragma unroll
  for (int e = 0; e < 8; ++e) a[8 + e] = p[8 + e];
  return a;
}

// 16 contiguous bf16 (fragments staged so they are contiguous in LDS)
__device__ __forceinline__ bf16x16 load_frag_contig(const __bf16* p) {
  bf16x16 v;
#pragma unroll
  for (int e = 0; e < 16; ++e) v[e] = p[e];
  return v;
}

__device__ __forceinline__ float rowmax16(float v) {
#pragma unroll
  for (int off = 8; off > 0; off >>= 1) v = fmaxf(v, __shfl_xor(v, off, 16));
  return v;
}
__device__ __forceinline__ float rowsum16(float v) {
#pragma unroll
  for (int off = 8; off > 0; off >>= 1) v += __shfl_xor(v, off, 16);
  return v;
}

// ---------------------------------------------------------------------------
// Weight prep: all GEMM B-operands are stored TRANSPOSED ([N][K], bf16) once,
// so GEMM tiles stage with contiguous 16B loads only.
// ---------------------------------------------------------------------------
// src: [K][N] f32 row-major  ->  dst: [N][K] bf16
__global__ void transpose_cast_bf16(const float* __restrict__ src,
                                    __bf16* __restrict__ dst, int N, int K) {
  int idx = blockIdx.x * blockDim.x + threadIdx.x;
  if (idx >= N * K) return;
  int n = idx / K;
  int k = idx % K;
  dst[idx] = (__bf16)src[(size_t)k * N + n];
}

// Wq/Wk/Wv: [H, C, HS] -> transposed packed [3C][C]; row n = sel*1024+h*64+s
__global__ void pack_qkv_weights_t(const float* __restrict__ Wq,
                                   const float* __restrict__ Wk,
                                   const float* __restrict__ Wv,
                                   __bf16* __restrict__ dst) {
  int idx = blockIdx.x * blockDim.x + threadIdx.x;
  if (idx >= LDQKV * Cv) return;
  int n = idx >> 10;          // 0..3071
  int c = idx & 1023;
  int sel = n >> 10;
  int w   = n & 1023;
  int hd  = w >> 6;
  int s   = w & 63;
  const float* W = (sel == 0) ? Wq : (sel == 1) ? Wk : Wv;
  dst[idx] = (__bf16)W[(size_t)hd * Cv * HSv + (size_t)c * HSv + s];
}

// ---------------------------------------------------------------------------
// LayerNorm: one block per row, C=1024, 256 threads x 4 elements
// ---------------------------------------------------------------------------
__global__ void layernorm_bf16(const float* __restrict__ x,
                               const float* __restrict__ w,
                               const float* __restrict__ bias,
                               __bf16* __restrict__ out) {
  const int row = blockIdx.x;
  const int t   = threadIdx.x;
  const float* xr = x + (size_t)row * Cv;
  float vals[4], s = 0.f, ss = 0.f;
#pragma unroll
  for (int i = 0; i < 4; ++i) {
    float v = xr[i * 256 + t];
    vals[i] = v; s += v; ss += v * v;
  }
  __shared__ float rs[256], rss[256];
  rs[t] = s; rss[t] = ss;
  __syncthreads();
  for (int off = 128; off > 0; off >>= 1) {
    if (t < off) { rs[t] += rs[t + off]; rss[t] += rss[t + off]; }
    __syncthreads();
  }
  const float inv = 1.0f / (float)Cv;
  float mu  = rs[0] * inv;
  float var = rss[0] * inv - mu * mu;
  float rstd = rsqrtf(var + 1e-5f);
  __bf16* orow = out + (size_t)row * Cv;
#pragma unroll
  for (int i = 0; i < 4; ++i) {
    int c = i * 256 + t;
    orow[c] = (__bf16)((vals[i] - mu) * rstd * w[c] + bias[c]);
  }
}

// ---------------------------------------------------------------------------
// Tiled bf16 WMMA GEMM, double-buffered with CDNA5 async global->LDS DMA.
//   Block = 256 threads = 8 waves -> 128x64 output tile.
//   Waves 4x2; each wave owns 32x32 (2x2 accumulators -> 4 WMMAs / k-step;
//   A- and B-fragments each reused twice).
//   B operand is pre-transposed [N][K]: every staged chunk is one 16B-aligned
//   global_load_async_to_lds_b128 (no VGPR round-trip); fragments are
//   contiguous ds_load_b128 pairs.
//   Pipeline: async copies for step i+1 issue before step-i WMMAs (DMA overlaps
//   matrix math), s_wait_asynccnt 0 + one barrier per k-step publishes the
//   alternate buffer.
// ---------------------------------------------------------------------------
template <int MODE>
__global__ void gemm_tile_wmma(const __bf16* __restrict__ A, int lda,
                               const __bf16* __restrict__ BT,   // [N][K]
                               const float* __restrict__ resid,
                               float* __restrict__ outF,
                               __bf16* __restrict__ outB,
                               int ldc, int K) {
  __shared__ __bf16 As[2][128 * 32];   // [row][k]
  __shared__ __bf16 Bs[2][64 * 32];    // [col][k]

  const int t    = threadIdx.x;
  const int lane = t & 31;
  const int wave = t >> 5;
  const int wrow = wave >> 1;          // 0..3 -> 32-row band
  const int wcol = wave & 1;           // 0..1 -> 32-col band
  const int tm   = blockIdx.y * 128;
  const int tn   = blockIdx.x * 64;
  const int n    = lane & 15;
  const int kg   = lane >> 4;

  // cooperative-load coordinates (contiguous 16B chunks per thread)
  const int ar  = t >> 2;              // 0..63 (A rows, +64 for 2nd chunk)
  const int ak  = (t & 3) * 8;         // 0,8,16,24
  const int bn  = t >> 2;              // 0..63 (B cols)
  const int bk  = (t & 3) * 8;

  const __bf16* Ag0 = A  + (size_t)(tm + ar) * lda + ak;        // chunk 0
  const __bf16* Ag1 = A  + (size_t)(tm + ar + 64) * lda + ak;   // chunk 1
  const __bf16* Bg  = BT + (size_t)(tn + bn) * K + bk;

  // per-thread LDS destinations for both buffers (byte offsets)
  uint32_t lA0[2], lA1[2], lB[2];
#pragma unroll
  for (int bufi = 0; bufi < 2; ++bufi) {
    lA0[bufi] = (uint32_t)(uintptr_t)&As[bufi][ar * 32 + ak];
    lA1[bufi] = (uint32_t)(uintptr_t)&As[bufi][(ar + 64) * 32 + ak];
    lB[bufi]  = (uint32_t)(uintptr_t)&Bs[bufi][bn * 32 + bk];
  }

  const int nsteps = K >> 5;

  // prologue: async-stage tile 0 into buffer 0
  async_ld_b128(lA0[0], Ag0);
  async_ld_b128(lA1[0], Ag1);
  async_ld_b128(lB[0],  Bg);
  wait_async0();
  __syncthreads();

  f32x8 acc[2][2];
#pragma unroll
  for (int mt = 0; mt < 2; ++mt)
#pragma unroll
    for (int nt = 0; nt < 2; ++nt) acc[mt][nt] = f32x8{};

  for (int i = 0; i < nsteps; ++i) {
    const int cur = i & 1;
    const int nxt = cur ^ 1;
    const bool has_next = (i + 1) < nsteps;

    // launch DMA for next tile; overlaps with the WMMAs below
    if (has_next) {
      const int ko = (i + 1) << 5;
      async_ld_b128(lA0[nxt], Ag0 + ko);
      async_ld_b128(lA1[nxt], Ag1 + ko);
      async_ld_b128(lB[nxt],  Bg + ko);
    }

    // compute from current buffer: 2 A-frags, 2 B-frags, 4 WMMAs
    bf16x16 af[2], bf[2];
#pragma unroll
    for (int mt = 0; mt < 2; ++mt)
      af[mt] = load_a_frag(As[cur] + (wrow * 32 + mt * 16) * 32, 32, lane);
#pragma unroll
    for (int nt = 0; nt < 2; ++nt)
      bf[nt] = load_frag_contig(Bs[cur] + (wcol * 32 + nt * 16 + n) * 32 + kg * 16);
#pragma unroll
    for (int mt = 0; mt < 2; ++mt)
#pragma unroll
      for (int nt = 0; nt < 2; ++nt)
        acc[mt][nt] = __builtin_amdgcn_wmma_f32_16x16x32_bf16(
            false, af[mt], false, bf[nt], (short)0, acc[mt][nt], false, false);

    if (has_next) wait_async0();       // this wave's DMA done
    __syncthreads();                    // publish buffer across waves
  }

  // epilogue (compile-time mode; no branches in the store loop)
  const int hi = lane >> 4;
#pragma unroll
  for (int mt = 0; mt < 2; ++mt) {
#pragma unroll
    for (int nt = 0; nt < 2; ++nt) {
#pragma unroll
      for (int r = 0; r < 8; ++r) {
        int row = tm + wrow * 32 + mt * 16 + r + hi * 8;
        int col = tn + wcol * 32 + nt * 16 + n;
        size_t idx = (size_t)row * ldc + col;
        float v = acc[mt][nt][r];
        if constexpr (MODE == MODE_F32_RESID) {
          outF[idx] = v + resid[idx];
        } else if constexpr (MODE == MODE_BF16_RELU) {
          outB[idx] = (__bf16)fmaxf(v, 0.0f);
        } else {
          outB[idx] = (__bf16)v;
        }
      }
    }
  }
}

// ---------------------------------------------------------------------------
// Causal flash attention.  qkv: [B*T, 3C] bf16 (cols: Q | K | V, each h*64+s).
// 1 wave = 16 query rows; 8 waves/block share (b,h) and 128 query rows so the
// key loop is block-uniform (causality bound).  32-key tiles staged in LDS,
// V transposed at stage time so the P*V B-fragments are contiguous ds loads.
// Masking only on diagonal-crossing tiles (wave-uniform test).
// ---------------------------------------------------------------------------
__global__ void attention_wmma(const __bf16* __restrict__ qkv,
                               const float* __restrict__ p,
                               __bf16* __restrict__ att) {
  __shared__ __bf16 Kt[32 * 64];     // [key][dim]
  __shared__ __bf16 VtT[64 * 32];    // [dim][key]
  __shared__ __bf16 Pt[8][16 * 32];  // per-wave P tile [m][k]

  const int nqb  = Tv >> 7;                       // 16 query blocks of 128
  const int qblk = blockIdx.x % nqb;
  const int h    = (blockIdx.x / nqb) % Hv;
  const int b    = blockIdx.x / (nqb * Hv);
  const int wave = threadIdx.x >> 5;
  const int lane = threadIdx.x & 31;
  const int row0 = qblk * 128 + wave * 16;        // first query row (local t)

  const size_t base = (size_t)b * Tv * LDQKV;
  const __bf16* Qb = qkv + base + (size_t)h * 64;
  const __bf16* Kb = qkv + base + Cv     + (size_t)h * 64;
  const __bf16* Vb = qkv + base + 2 * Cv + (size_t)h * 64;

  // Q tile (16x64) kept in registers as two K=32 A-fragments
  bf16x16 qa0 = load_a_frag(Qb + (size_t)row0 * LDQKV,      LDQKV, lane);
  bf16x16 qa1 = load_a_frag(Qb + (size_t)row0 * LDQKV + 32, LDQKV, lane);

  const float scale = rsqrtf(p[h]);
  const int n  = lane & 15;
  const int hi = lane >> 4;

  float m_i[8], l_i[8];
#pragma unroll
  for (int r = 0; r < 8; ++r) { m_i[r] = -INFINITY; l_i[r] = 0.0f; }
  f32x8 o[4];
#pragma unroll
  for (int d = 0; d < 4; ++d) o[d] = f32x8{};

  // cooperative-load coordinates: 256 threads x 8 bf16 = one 32x64 tile
  const int le   = threadIdx.x * 8;
  const int krow = le >> 6;
  const int kcol = le & 63;

  const int nkt = (qblk + 1) * 4;                 // causal bound, 32-key tiles
  for (int kt = 0; kt < nkt; ++kt) {
    const int key0 = kt * 32;
    // stage K (async DMA, row-major) and V (manual transpose) in LDS
    async_ld_b128((uint32_t)(uintptr_t)&Kt[krow * 64 + kcol],
                  Kb + (size_t)(key0 + krow) * LDQKV + kcol);
#pragma unroll
    for (int i = 0; i < 8; ++i) {
      VtT[(kcol + i) * 32 + krow] =
          Vb[(size_t)(key0 + krow) * LDQKV + kcol + i];
    }
    wait_async0();
    __syncthreads();

    const bool needMask = (key0 + 31) > row0;     // wave-uniform

    // S = Q K^T for 2 n-tiles of 16 keys (K-dim = HS = 64 -> 2 WMMAs each)
    float sv0[8], sv1[8];
#pragma unroll
    for (int ntile = 0; ntile < 2; ++ntile) {
      f32x8 s{};
      bf16x16 b0 = load_frag_contig(Kt + (ntile * 16 + n) * 64 +      hi * 16);
      s = __builtin_amdgcn_wmma_f32_16x16x32_bf16(false, qa0, false, b0,
                                                  (short)0, s, false, false);
      bf16x16 b1 = load_frag_contig(Kt + (ntile * 16 + n) * 64 + 32 + hi * 16);
      s = __builtin_amdgcn_wmma_f32_16x16x32_bf16(false, qa1, false, b1,
                                                  (short)0, s, false, false);
      float* sv = ntile ? sv1 : sv0;
      if (needMask) {
        const int colg = key0 + ntile * 16 + n;
#pragma unroll
        for (int r = 0; r < 8; ++r) {
          int rowg = row0 + r + hi * 8;
          float v = s[r] * scale;
          sv[r] = (colg > rowg) ? -INFINITY : v;
        }
      } else {
#pragma unroll
        for (int r = 0; r < 8; ++r) sv[r] = s[r] * scale;
      }
    }

    // online softmax update + write P tile to LDS
#pragma unroll
    for (int r = 0; r < 8; ++r) {
      float mx = rowmax16(fmaxf(sv0[r], sv1[r]));
      float mn = fmaxf(m_i[r], mx);
      float alpha = __expf(m_i[r] - mn);
      float p0 = __expf(sv0[r] - mn);
      float p1 = __expf(sv1[r] - mn);
      float rsum = rowsum16(p0 + p1);
      l_i[r] = l_i[r] * alpha + rsum;
      m_i[r] = mn;
#pragma unroll
      for (int d = 0; d < 4; ++d) o[d][r] *= alpha;
      int m = r + hi * 8;
      Pt[wave][m * 32 + n]      = (__bf16)p0;
      Pt[wave][m * 32 + 16 + n] = (__bf16)p1;
    }
    __syncthreads();

    // O += P (16x32) * V (32x64): one A-frag of P, 4 B-frags of V^T
    bf16x16 pa = load_a_frag(&Pt[wave][0], 32, lane);
#pragma unroll
    for (int d = 0; d < 4; ++d) {
      bf16x16 vb = load_frag_contig(VtT + (d * 16 + n) * 32 + hi * 16);
      o[d] = __builtin_amdgcn_wmma_f32_16x16x32_bf16(false, pa, false, vb,
                                                     (short)0, o[d], false, false);
    }
    __syncthreads();   // protect Kt/VtT/Pt before next iteration
  }

  // normalize and write [B*T, C] bf16 (heads concatenated)
#pragma unroll
  for (int r = 0; r < 8; ++r) {
    int m = r + hi * 8;
    float inv = 1.0f / l_i[r];
    size_t orow = (size_t)(b * Tv + row0 + m) * Cv + (size_t)h * 64;
#pragma unroll
    for (int d = 0; d < 4; ++d)
      att[orow + d * 16 + n] = (__bf16)(o[d][r] * inv);
  }
}

// ---------------------------------------------------------------------------
// Launch
// ---------------------------------------------------------------------------
extern "C" void kernel_launch(void* const* d_in, const int* in_sizes, int n_in,
                              void* d_out, int out_size, void* d_ws, size_t ws_size,
                              hipStream_t stream) {
  const float* x     = (const float*)d_in[0];
  const float* Wq    = (const float*)d_in[1];
  const float* Wk    = (const float*)d_in[2];
  const float* Wv    = (const float*)d_in[3];
  const float* p     = (const float*)d_in[4];
  const float* Wproj = (const float*)d_in[5];
  const float* W1    = (const float*)d_in[6];
  const float* W2    = (const float*)d_in[7];
  const float* ln1w  = (const float*)d_in[8];
  const float* ln1b  = (const float*)d_in[9];
  const float* ln2w  = (const float*)d_in[10];
  const float* ln2b  = (const float*)d_in[11];
  float* out = (float*)d_out;

  char* ws = (char*)d_ws;
  size_t off = 0;
  auto alloc = [&](size_t bytes) { char* r = ws + off; off += (bytes + 255) & ~(size_t)255; return r; };
  __bf16* wqkvT  = (__bf16*)alloc((size_t)LDQKV * Cv * 2);      // [3C][C]   6 MB
  __bf16* wprojT = (__bf16*)alloc((size_t)Cv * Cv * 2);         // [C][C]    2 MB
  __bf16* w1T    = (__bf16*)alloc((size_t)FFv * Cv * 2);        // [FF][C]   8 MB
  __bf16* w2T    = (__bf16*)alloc((size_t)Cv * FFv * 2);        // [C][FF]   8 MB
  __bf16* hbuf   = (__bf16*)alloc((size_t)M_ROWS * Cv * 2);     // 8 MB
  __bf16* qkvb   = (__bf16*)alloc((size_t)M_ROWS * LDQKV * 2);  // 24 MB
  __bf16* attb   = (__bf16*)alloc((size_t)M_ROWS * Cv * 2);     // 8 MB
  float*  x1f    = (float*) alloc((size_t)M_ROWS * Cv * 4);     // 16 MB
  __bf16* h2b    = (__bf16*)alloc((size_t)M_ROWS * Cv * 2);     // 8 MB
  __bf16* ffb    = (__bf16*)alloc((size_t)M_ROWS * FFv * 2);    // 32 MB

  // 1. weight prep (transposed bf16, one-time per launch)
  pack_qkv_weights_t<<<(LDQKV * Cv + 255) / 256, 256, 0, stream>>>(Wq, Wk, Wv, wqkvT);
  transpose_cast_bf16<<<(Cv * Cv + 255) / 256, 256, 0, stream>>>(Wproj, wprojT, Cv, Cv);
  transpose_cast_bf16<<<(FFv * Cv + 255) / 256, 256, 0, stream>>>(W1, w1T, FFv, Cv);
  transpose_cast_bf16<<<(Cv * FFv + 255) / 256, 256, 0, stream>>>(W2, w2T, Cv, FFv);

  // 2. LN1
  layernorm_bf16<<<M_ROWS, 256, 0, stream>>>(x, ln1w, ln1b, hbuf);

  // 3. fused QKV GEMM: [4096,1024] x [1024,3072] -> bf16
  gemm_tile_wmma<MODE_BF16><<<dim3(LDQKV / 64, M_ROWS / 128), 256, 0, stream>>>(
      hbuf, Cv, wqkvT, nullptr, nullptr, qkvb, LDQKV, Cv);

  // 4. causal attention
  attention_wmma<<<Bv * Hv * (Tv / 128), 256, 0, stream>>>(qkvb, p, attb);

  // 5. output projection + residual -> x1 (f32)
  gemm_tile_wmma<MODE_F32_RESID><<<dim3(Cv / 64, M_ROWS / 128), 256, 0, stream>>>(
      attb, Cv, wprojT, x, x1f, nullptr, Cv, Cv);

  // 6. LN2
  layernorm_bf16<<<M_ROWS, 256, 0, stream>>>(x1f, ln2w, ln2b, h2b);

  // 7. FFN GEMM1 + ReLU -> bf16
  gemm_tile_wmma<MODE_BF16_RELU><<<dim3(FFv / 64, M_ROWS / 128), 256, 0, stream>>>(
      h2b, Cv, w1T, nullptr, nullptr, ffb, FFv, Cv);

  // 8. FFN GEMM2 + residual -> d_out (f32)
  gemm_tile_wmma<MODE_F32_RESID><<<dim3(Cv / 64, M_ROWS / 128), 256, 0, stream>>>(
      ffb, FFv, w2T, x1f, out, nullptr, Cv, FFv);
}
